// stableKT_15436112462208
// MI455X (gfx1250) — matrix operand order repaired and checked by reading, not compiled, not run on hardware
//
#include <hip/hip_runtime.h>
#include <math.h>

#define B_ 4
#define H_ 8
#define S_ 2048
#define D_ 64
#define GAMMA_ 0.1f
#define EPS_ 1e-6f
#define SCALE_ 0.125f      /* 1/sqrt(64) */
#define NEGINF_ -3.0e38f

typedef _Float16 v16h __attribute__((ext_vector_type(16)));
typedef float    v8f  __attribute__((ext_vector_type(8)));

__device__ __forceinline__ v8f wmma_f16(v16h a, v16h b, v8f c) {
  // D = A(16x32 f16) * B(32x16 f16) + C(16x16 f32)
  return __builtin_amdgcn_wmma_f32_16x16x32_f16(false, a, false, b, (short)0, c, false, false);
}

__device__ __forceinline__ float sigmoidf_(float x) {
  return 1.0f / (1.0f + __expf(-x));
}

__global__ __launch_bounds__(128)
void penumbral_flash_kernel(const float* __restrict__ q,
                            const float* __restrict__ k,
                            const float* __restrict__ v,
                            const float* __restrict__ alibi,
                            float* __restrict__ out) {
  __shared__ __align__(16) float   s_qraw[64 * 64];   // 16 KB raw Q block
  __shared__ float4                 s_qscal[64];       // (q_sq, q_y^2, x_qy, q_y)
  __shared__ __align__(16) float   s_kraw[32 * 64];   // 8 KB raw K tile
  __shared__ __align__(16) float   s_vraw[32 * 64];   // 8 KB raw V tile
  __shared__ float4                 s_kscal[32];       // (k_sq, k_y^2, x_ky, k_y)
  __shared__ __align__(32) _Float16 s_ak[4 * 32 * 16]; // A-layout f16 (-2*k_x), 4 chunks
  __shared__ __align__(32) _Float16 s_bv[4 * 32 * 16]; // B-layout f16 V, 4 d-chunks

  const int tid   = threadIdx.x;
  const int wave  = tid >> 5;
  const int lane  = tid & 31;
  const int qbase = blockIdx.x * 64;
  const int bh    = blockIdx.y;            // b*H + h
  const int h     = bh & (H_ - 1);

  const size_t base = (size_t)bh * S_ * D_;
  const float* qp = q + base;
  const float* kp = k + base;
  const float* vp = v + base;
  float*       op = out + base;

  // alibi[0,h,1,0] = slope_h * rel(1,0) = -slope_h  ->  exact slope recovery
  const float slope = -alibi[(size_t)h * S_ * S_ + S_];

  // ---------------- Phase 0: stage Q block, per-query scalars ----------------
  {
    const float4* g  = (const float4*)(qp + (size_t)qbase * D_);
    float4*       sh = (float4*)s_qraw;
#pragma unroll
    for (int u = 0; u < 8; ++u) sh[tid + 128 * u] = g[tid + 128 * u];
  }
  __syncthreads();
  if (tid < 64) {
    const float* row = &s_qraw[tid * 64];
    float qy = sigmoidf_(row[63]);
    float ss = 0.f;
    for (int d = 0; d < 63; ++d) ss = fmaf(row[d], row[d], ss);
    float qy2 = qy * qy;
    s_qscal[tid] = make_float4(ss * qy2, qy2, sqrtf(1.0f - qy2 + EPS_), qy);
  }
  __syncthreads();

  const int   ql  = wave * 16 + (lane & 15);   // local query row of this lane
  const int   hi8 = (lane & 16) ? 8 : 0;
  const float4 qv = s_qscal[ql];
  const float q_sq = qv.x, q_y2 = qv.y, x_qy = qv.z, q_y = qv.w;

  // Build resident B operands for q_x (features 0..62, dim 63 = 0), B layout:
  // lane = query(N); K=c*32+(j<4?2j:16+2(j-4)), +8 for lanes 16-31
  v16h bq0, bq1;
  {
    const float* row = &s_qraw[ql * 64];
    const int fb = (lane & 16) ? 8 : 0;
#pragma unroll
    for (int j = 0; j < 8; ++j) {
      const int f0 = fb + ((j < 4) ? 2 * j : 16 + 2 * (j - 4));
      const int f1 = 32 + fb + ((j < 4) ? 2 * j : 16 + 2 * (j - 4));
      bq0[2 * j]     = (_Float16)(row[f0] * q_y);
      bq0[2 * j + 1] = (_Float16)(row[f0 + 1] * q_y);
      bq1[2 * j]     = (_Float16)(row[f1] * q_y);
      bq1[2 * j + 1] = (_Float16)((f1 + 1 < 63) ? row[f1 + 1] * q_y : 0.f);
    }
  }

  float m_i = NEGINF_;
  float l_i = 0.f;
  v8f O0 = {}, O1 = {}, O2 = {}, O3 = {};

  const int ntiles = (qbase + 64) / 32;             // causal tile count
  const int qmax_w = qbase + wave * 16 + 15;        // wave's largest query
  const int i_g    = qbase + ql;                    // this lane's query index

  const v16h* pak = (const v16h*)s_ak;
  const v16h* pbv = (const v16h*)s_bv;

  for (int kt = 0; kt < ntiles; ++kt) {
    const int ktbase = kt * 32;
    __syncthreads();                                // prior consumers done
    // ---- coalesced raw staging of K/V tile ----
    {
      const float4* gk = (const float4*)(kp + (size_t)ktbase * D_);
      const float4* gv = (const float4*)(vp + (size_t)ktbase * D_);
      float4* shk = (float4*)s_kraw;
      float4* shv = (float4*)s_vraw;
#pragma unroll
      for (int u = 0; u < 4; ++u) {
        shk[tid + 128 * u] = gk[tid + 128 * u];
        shv[tid + 128 * u] = gv[tid + 128 * u];
      }
      if (kt + 1 < ntiles) {                        // global_prefetch_b8
        __builtin_prefetch(kp + (size_t)(ktbase + 32) * D_ + tid * 4, 0, 1);
        __builtin_prefetch(vp + (size_t)(ktbase + 32) * D_ + tid * 4, 0, 1);
      }
    }
    __syncthreads();
    // ---- cooperative pack: A-layout (-2*k_x) f16, B-layout V f16, k-scalars ----
    {
      const int a   = tid >> 5;                     // chunk 0..3
      const int l   = tid & 31;
      const int s   = a >> 1;                       // key subtile
      const int c   = a & 1;                        // feature chunk
      const int key = s * 16 + (l & 15);
      const int h8  = (l & 16) ? 8 : 0;
      const float km = -2.0f * sigmoidf_(s_kraw[key * 64 + 63]);
      _Float16* dst = &s_ak[a * 512 + l * 16];
#pragma unroll
      for (int j = 0; j < 8; ++j) {                 // A K-striping: 2j | 16+2(j-4), +8 upper half
        const int f = c * 32 + ((j < 4) ? 2 * j : 16 + 2 * (j - 4)) + h8;
        dst[2 * j]     = (_Float16)(s_kraw[key * 64 + f] * km);
        dst[2 * j + 1] = (_Float16)((f + 1 < 63) ? s_kraw[key * 64 + f + 1] * km : 0.f);
      }
      const int d  = a * 16 + (l & 15);             // B-layout V: lane = d col
      const int kb = (l & 16) ? 8 : 0;
      _Float16* dv = &s_bv[a * 512 + l * 16];
#pragma unroll
      for (int j = 0; j < 8; ++j) {
        dv[2 * j]     = (_Float16)s_vraw[(kb + ((j < 4) ? 2 * j : 16 + 2 * (j - 4))) * 64 + d];
        dv[2 * j + 1] = (_Float16)s_vraw[(kb + ((j < 4) ? 2 * j : 16 + 2 * (j - 4)) + 1) * 64 + d];
      }
      if (tid < 32) {
        const float* row = &s_kraw[tid * 64];
        float ky = sigmoidf_(row[63]);
        float ky2 = ky * ky;
        float ss = 0.f;
        for (int dd = 0; dd < 63; ++dd) ss = fmaf(row[dd], row[dd], ss);
        s_kscal[tid] = make_float4(ss * ky2, ky2, sqrtf(1.0f - ky2 + EPS_), ky);
      }
    }
    __syncthreads();

    if (ktbase > qmax_w) continue;                  // wave-uniform skip (fully masked)

    // ---- score tiles: C[key, query] = (-2 k_x) . q_x, f32 accumulate over 64 dims ----
    v8f c0 = {}, c1 = {};
    c0 = wmma_f16(pak[0 * 32 + lane], bq0, c0);
    c0 = wmma_f16(pak[1 * 32 + lane], bq1, c0);
    c1 = wmma_f16(pak[2 * 32 + lane], bq0, c1);
    c1 = wmma_f16(pak[3 * 32 + lane], bq1, c1);

    // ---- penumbral score + ALiBi + causal mask, per element ----
    float sc[16];
    float tmax = NEGINF_;
#pragma unroll
    for (int e = 0; e < 16; ++e) {
      const int s     = e >> 3;
      const int rr    = e & 7;
      const int t_loc = s * 16 + rr + hi8;
      const int t_g   = ktbase + t_loc;
      const float dotm2 = (s == 0) ? c0[rr] : c1[rr];
      const float4 kv = s_kscal[t_loc];
      float sq  = q_sq + kv.x + dotm2;              // |q|^2+|k|^2-2qk
      float pd2 = fmaxf(sq, 0.f);
      float pd  = sqrtf(pd2);
      float tt  = (x_qy + kv.z - pd) * 0.5f;
      float lh  = fmaxf(fmaxf(q_y2, kv.y), 1.0f - tt * tt);
      float nm  = pd2 + kv.y - q_y2;
      float qq  = nm / (2.0f * pd + EPS_);
      float lo  = fmaf(qq, qq, q_y2);
      float dxa = pd - x_qy;
      bool cone = (pd <= x_qy) || (fmaf(dxa, dxa, kv.y) <= 1.0f);
      float pen = -GAMMA_ * (cone ? lh : lo);
      float val = fmaf(pen, SCALE_, slope * (float)(t_g - i_g));
      sc[e] = (t_g <= i_g) ? val : NEGINF_;
      tmax  = fmaxf(tmax, sc[e]);
    }
    tmax = fmaxf(tmax, __shfl_xor(tmax, 16));       // combine lane halves (same query)
    const float m_new = fmaxf(m_i, tmax);
    const float alpha = __expf(m_i - m_new);
    float lsum = 0.f;
#pragma unroll
    for (int e = 0; e < 16; ++e) {
      float p = __expf(sc[e] - m_new);
      sc[e] = p;
      lsum += p;
    }
    lsum += __shfl_xor(lsum, 16);
    l_i = fmaf(l_i, alpha, lsum);
    m_i = m_new;

    // ---- rescale O (per-query broadcast to C-row layout) ----
    float ab[8];
#pragma unroll
    for (int r2 = 0; r2 < 8; ++r2) ab[r2] = __shfl(alpha, r2 + hi8);
#pragma unroll
    for (int r2 = 0; r2 < 8; ++r2) {
      O0[r2] *= ab[r2]; O1[r2] *= ab[r2]; O2[r2] *= ab[r2]; O3[r2] *= ab[r2];
    }

    // ---- P tile -> A operand: pure lane-local cvt/pack (C layout == A layout here) ----
    v16h ap;
#pragma unroll
    for (int e = 0; e < 8; ++e) {
      ap[e]     = (_Float16)sc[e];
      ap[e + 8] = (_Float16)sc[e + 8];
    }
    O0 = wmma_f16(ap, pbv[0 * 32 + lane], O0);
    O1 = wmma_f16(ap, pbv[1 * 32 + lane], O1);
    O2 = wmma_f16(ap, pbv[2 * 32 + lane], O2);
    O3 = wmma_f16(ap, pbv[3 * 32 + lane], O3);
  }

  // ---------------- Epilogue: normalize, ZERO_PAD row 0, store ----------------
  const float linv = 1.0f / l_i;
  float lb[8];
#pragma unroll
  for (int r2 = 0; r2 < 8; ++r2) lb[r2] = __shfl(linv, r2 + hi8);
  const int dcol = lane & 15;
#pragma unroll
  for (int r2 = 0; r2 < 8; ++r2) {
    const int qg = qbase + wave * 16 + r2 + hi8;
    const float z = (qg == 0) ? 0.f : 1.f;          // ZERO_PAD: attn row 0 -> output 0
    float* orow = op + (size_t)qg * 64 + dcol;
    orow[0]  = O0[r2] * lb[r2] * z;
    orow[16] = O1[r2] * lb[r2] * z;
    orow[32] = O2[r2] * lb[r2] * z;
    orow[48] = O3[r2] * lb[r2] * z;
  }
}

extern "C" void kernel_launch(void* const* d_in, const int* in_sizes, int n_in,
                              void* d_out, int out_size, void* d_ws, size_t ws_size,
                              hipStream_t stream) {
  (void)in_sizes; (void)n_in; (void)out_size; (void)d_ws; (void)ws_size;
  const float* q     = (const float*)d_in[0];
  const float* k     = (const float*)d_in[1];
  const float* v     = (const float*)d_in[2];
  const float* alibi = (const float*)d_in[3];
  // d_in[4] (mask) is the causal tril; computed analytically in-kernel.
  float* out = (float*)d_out;

  dim3 grid(S_ / 64, B_ * H_);
  penumbral_flash_kernel<<<grid, 128, 0, stream>>>(q, k, v, alibi, out);
}
